// ScaledDotProductAttention_2001454760328
// MI455X (gfx1250) — matrix operand order repaired and checked by reading, not compiled, not run on hardware
//
#include <hip/hip_runtime.h>

// ---------------------------------------------------------------------------
// ScaledDotProductAttention forward (training-mode dropout), MI455X / gfx1250.
//   scores = Q K^T / sqrt(64) ; attn = dropout(softmax(scores)) ; y = attn V
// Outputs (concatenated in d_out): y [B,S,64] f32, attn [B,S,S] f32.
// One workgroup (8 wave32) per (batch, 16-query-row tile). The full 16x2048
// score strip lives in LDS; V is double-buffered through LDS in 32KB chunks
// with CDNA5 async global->LDS copies (ASYNCcnt), chunk 0 issued at kernel
// entry so its latency hides behind phases 1-2. Both GEMMs use
// v_wmma_f32_16x16x32_f16. attn/mask streams use non-temporal hints so the
// 512MB attn write does not evict L2-resident Q/K/V (48MB << 192MB L2).
// ---------------------------------------------------------------------------

#define BATCH 32
#define SEQ   2048
#define DK    64
#define NTILES (SEQ / 16)          // 128 key tiles per row strip
#define SCALE 0.125f               // 1/sqrt(64), folded into Q->f16 convert
#define KEEP_SCALE 2.0f            // 1/(1-p), p=0.5
#define VCHUNK_ROWS  128           // rows of V per async-staged chunk
#define VCHUNK_BYTES (VCHUNK_ROWS * DK * 4)   // 32KB
#define NCHUNK (SEQ / VCHUNK_ROWS)            // 16

typedef __attribute__((ext_vector_type(16))) _Float16 v16h;
typedef __attribute__((ext_vector_type(8)))  float    v8f;
typedef __attribute__((ext_vector_type(4)))  float    v4f;

// LDS layout (dynamic, base offset 0 since no static __shared__):
//   sc   [16][2048] f32     scores -> exp values          131072 B @ 0
//   af   [16][2048] f16     post-dropout attn (GEMM2 A)    65536 B @ 131072
//   red  [16][16]   f32     softmax reduction scratch       1024 B @ 196608
//   par  [4][16][16]f32     GEMM2 K-half partials           4096 B @ 197632
//   vbuf [2][128][64] f32   async-staged V double buffer   65536 B @ 201728
#define AF_OFF   (16 * SEQ * 4)
#define RED_OFF  (AF_OFF + 16 * SEQ * 2)
#define PAR_OFF  (RED_OFF + 16 * 16 * 4)
#define VBUF_OFF (PAR_OFF + 4 * 16 * 16 * 4)          // 201728
#define SMEM_BYTES (VBUF_OFF + 2 * VCHUNK_BYTES)      // 267264 (<320KB/WGP)

// Async-copy one 32KB V chunk into LDS buffer bufsel (0/1).
// 256 threads x 8 x b128: thread t moves flat floats [t*4 + i*1024].
__device__ __forceinline__ void issue_vchunk_copy(const float* __restrict__ V,
                                                  int b, int ch, int bufsel,
                                                  int tid) {
    const float* gsrc = V + ((size_t)(b * SEQ + ch * VCHUNK_ROWS)) * DK + tid * 4;
    const unsigned int ldsbase = (unsigned int)VBUF_OFF
                               + (unsigned int)bufsel * (unsigned int)VCHUNK_BYTES
                               + (unsigned int)tid * 16u;
    #pragma unroll
    for (int i = 0; i < 8; ++i) {
        unsigned int       ldsa = ldsbase + (unsigned int)i * 4096u;
        unsigned long long ga   = (unsigned long long)(gsrc + i * 1024);
        asm volatile("global_load_async_to_lds_b128 %0, %1, off"
                     :: "v"(ldsa), "v"(ga) : "memory");
    }
}

__global__ __launch_bounds__(256)
void sdpa_fwd_kernel(const float* __restrict__ Q,
                     const float* __restrict__ Kmat,
                     const float* __restrict__ V,
                     const unsigned char* __restrict__ mask,
                     float* __restrict__ y,
                     float* __restrict__ attn)
{
    extern __shared__ char smem_raw[];
    float*    sc   = (float*)smem_raw;                    // [16][SEQ]
    _Float16* af   = (_Float16*)(smem_raw + AF_OFF);      // [16][SEQ]
    float*    red  = (float*)(smem_raw + RED_OFF);        // [16][16]
    float*    par  = (float*)(smem_raw + PAR_OFF);        // [4][16][16]
    float*    vbuf = (float*)(smem_raw + VBUF_OFF);       // [2][128][64]

    const int tid  = threadIdx.x;
    const int lane = tid & 31;
    const int wave = tid >> 5;
    const int b    = blockIdx.x >> 7;     // blockIdx.x / 128
    const int qt   = blockIdx.x & 127;    // query tile within sequence

    const int half = lane >> 4;           // half-wave (K-split in A/B layouts)
    const int l16  = lane & 15;           // row (A) / column (B,C,D) index

    // Kick off V chunk 0 now: its latency hides behind phases 1-2 entirely.
    issue_vchunk_copy(V, b, 0, 0, tid);

    // ===== Phase 1: scores = (Q * scale) K^T ===============================
    // A operand: 16x64 Q tile as two 16x32 f16 slabs; scale folded in here.
    // A k order used here (contiguous, matches B): lanes 0-15 row=l hold K {0..15}; lanes 16-31
    // same rows hold K {16..31}.
    const float* qrow = Q + ((size_t)(b * SEQ + qt * 16 + l16)) * DK;
    v16h a0, a1;
    #pragma unroll
    for (int j = 0; j < 8; ++j) {
        a0[j]     = (_Float16)(qrow[half * 16 + j] * SCALE);
        a0[j + 8] = (_Float16)(qrow[8 + half * 16 + j] * SCALE);
        a1[j]     = (_Float16)(qrow[32 + half * 16 + j] * SCALE);
        a1[j + 8] = (_Float16)(qrow[40 + half * 16 + j] * SCALE);
    }

    for (int nt = wave; nt < NTILES; nt += 8) {
        // B operand: column n = key row nt*16+l16; half-wave picks K 0..15/16..31.
        const float* krow = Kmat + ((size_t)(b * SEQ + nt * 16 + l16)) * DK + half * 16;
        // branchless prefetch of the next tile this wave will consume
        const int ntp = (nt + 8 < NTILES) ? (nt + 8) : nt;
        __builtin_prefetch(Kmat + ((size_t)(b * SEQ + ntp * 16 + l16)) * DK + half * 16, 0, 1);
        v16h b0, b1;
        #pragma unroll
        for (int j = 0; j < 16; ++j) {
            b0[j] = (_Float16)krow[j];        // K-dim 0..31
            b1[j] = (_Float16)krow[32 + j];   // K-dim 32..63
        }
        v8f acc = {};
        acc = __builtin_amdgcn_wmma_f32_16x16x32_f16(false, a0, false, b0,
                                                     (short)0, acc, false, false);
        acc = __builtin_amdgcn_wmma_f32_16x16x32_f16(false, a1, false, b1,
                                                     (short)0, acc, false, false);
        // C/D layout: vgpr r, lane -> (m = r + half*8, n = l16)
        #pragma unroll
        for (int r = 0; r < 8; ++r)
            sc[(r + half * 8) * SEQ + nt * 16 + l16] = acc[r];
    }
    __syncthreads();

    // ===== Phase 2: softmax + dropout over each 2048-wide row ==============
    {
        const int row = tid >> 4;        // 0..15
        const int seg = tid & 15;        // 16 threads per row, 128 cols each
        const int c0  = seg * 128;
        float* srow = sc + row * SEQ;

        float mx = -3.402823466e38f;
        #pragma unroll 4
        for (int c = 0; c < 128; ++c) mx = fmaxf(mx, srow[c0 + c]);
        red[row * 16 + seg] = mx;
        __syncthreads();
        float rmax = -3.402823466e38f;
        #pragma unroll
        for (int i = 0; i < 16; ++i) rmax = fmaxf(rmax, red[row * 16 + i]);

        float sum = 0.0f;
        #pragma unroll 4
        for (int c = 0; c < 128; ++c) {
            float e = __expf(srow[c0 + c] - rmax);
            srow[c0 + c] = e;
            sum += e;
        }
        __syncthreads();                 // all reads of max partials done
        red[row * 16 + seg] = sum;
        __syncthreads();
        float rsum = 0.0f;
        #pragma unroll
        for (int i = 0; i < 16; ++i) rsum += red[row * 16 + i];
        const float inv = KEEP_SCALE / rsum;   // fold dropout keep-scale in

        const int qg = qt * 16 + row;
        // Streaming 512MB attn out + 128MB mask in: non-temporal, don't
        // pollute L2 (Q/K/V stay resident there for the other 127 blocks).
        const unsigned int* m32 =
            (const unsigned int*)(mask + ((size_t)(b * SEQ + qg)) * SEQ + c0);
        v4f*      arow = (v4f*)(attn + ((size_t)(b * SEQ + qg)) * SEQ + c0);
        _Float16* frow = af + row * SEQ + c0;
        #pragma unroll 2
        for (int c = 0; c < 128; c += 4) {
            unsigned int mm = __builtin_nontemporal_load(m32 + (c >> 2));
            v4f o;
            o.x = (mm & 0x000000ffu) ? srow[c0 + c + 0] * inv : 0.0f;
            o.y = (mm & 0x0000ff00u) ? srow[c0 + c + 1] * inv : 0.0f;
            o.z = (mm & 0x00ff0000u) ? srow[c0 + c + 2] * inv : 0.0f;
            o.w = (mm & 0xff000000u) ? srow[c0 + c + 3] * inv : 0.0f;
            __builtin_nontemporal_store(o, arow + (c >> 2));  // f32 attn output
            frow[c + 0] = (_Float16)o.x;       // f16 copy for GEMM 2
            frow[c + 1] = (_Float16)o.y;
            frow[c + 2] = (_Float16)o.z;
            frow[c + 3] = (_Float16)o.w;
        }
    }
    __syncthreads();

    // ===== Phase 3: y = attn (16x2048 f16, LDS) x V (2048x64) ==============
    // V double-buffered through LDS in 16 chunks of 128 rows via async
    // global->LDS copies; chunk i+1 streams in behind chunk i's WMMAs.
    // 8 waves = 4 output N-tiles x 2 row-halves; 2 WMMA K-steps per chunk.
    {
        const int ntile = wave & 3;
        const int sub   = wave >> 2;          // chunk row-half 0/1
        const _Float16* afm = af + l16 * SEQ; // A row m = l16
        v8f acc = {};

        for (int ch = 0; ch < NCHUNK; ++ch) {
            if (ch + 1 < NCHUNK) {
                issue_vchunk_copy(V, b, ch + 1, (ch + 1) & 1, tid);
                // in-order completion: <=8 outstanding => chunk ch landed
                asm volatile("s_wait_asynccnt 0x8" ::: "memory");
            } else {
                asm volatile("s_wait_asynccnt 0x0" ::: "memory");
            }
            __syncthreads();                  // chunk ch visible to all waves

            const float* buf = vbuf + (ch & 1) * (VCHUNK_BYTES / 4);
            #pragma unroll
            for (int s2 = 0; s2 < 2; ++s2) {
                const int kb = ch * VCHUNK_ROWS + sub * 64 + s2 * 32; // global k
                const int kl = sub * 64 + s2 * 32;                    // k in chunk
                v16h A;
                const int abase = kb + half * 16;
                #pragma unroll
                for (int j = 0; j < 8; ++j) {
                    A[j]     = afm[abase + j];
                    A[j + 8] = afm[abase + 8 + j];
                }
                // B operand: V rows kl..kl+31 (in LDS), cols ntile*16..+15
                const float* vrow = buf + (kl + half * 16) * DK + ntile * 16 + l16;
                v16h Bv;
                #pragma unroll
                for (int j = 0; j < 16; ++j) Bv[j] = (_Float16)vrow[j * DK];
                acc = __builtin_amdgcn_wmma_f32_16x16x32_f16(false, A, false, Bv,
                                                             (short)0, acc, false, false);
            }
            __syncthreads();                  // done reading before overwrite
        }

        if (sub == 1) {
            #pragma unroll
            for (int r = 0; r < 8; ++r)
                par[ntile * 256 + (r + half * 8) * 16 + l16] = acc[r];
        }
        __syncthreads();
        if (sub == 0) {
            #pragma unroll
            for (int r = 0; r < 8; ++r) {
                float v = acc[r] + par[ntile * 256 + (r + half * 8) * 16 + l16];
                y[((size_t)(b * SEQ + qt * 16 + r + half * 8)) * DK
                  + ntile * 16 + l16] = v;
            }
        }
    }
}

extern "C" void kernel_launch(void* const* d_in, const int* in_sizes, int n_in,
                              void* d_out, int out_size, void* d_ws, size_t ws_size,
                              hipStream_t stream) {
    const float*         Q = (const float*)d_in[0];
    const float*         K = (const float*)d_in[1];
    const float*         V = (const float*)d_in[2];
    const unsigned char* M = (const unsigned char*)d_in[3];  // jax bool, 1B/elem

    float* y    = (float*)d_out;                                   // [B,S,64]
    float* attn = (float*)d_out + (size_t)BATCH * SEQ * DK;        // [B,S,S]

    (void)in_sizes; (void)n_in; (void)out_size; (void)d_ws; (void)ws_size;

    hipError_t attr_rc = hipFuncSetAttribute(
        (const void*)sdpa_fwd_kernel,
        hipFuncAttributeMaxDynamicSharedMemorySize, SMEM_BYTES);
    (void)attr_rc;

    dim3 grid(BATCH * NTILES);   // 4096 workgroups
    dim3 block(256);             // 8 wave32
    sdpa_fwd_kernel<<<grid, block, SMEM_BYTES, stream>>>(Q, K, V, M, y, attn);
}
